// BaseModel_40106404610740
// MI455X (gfx1250) — matrix operand order repaired and checked by reading, not compiled, not run on hardware
//
#include <hip/hip_runtime.h>
#include <hip/hip_bf16.h>
#include <math.h>

// ---------------------------------------------------------------------------
// Model dims (match reference)
//   B=2048, T=128, I=256, O=8, W=16, K=32, L=256, GN=512, H=512, DI=512
// ---------------------------------------------------------------------------

typedef __attribute__((ext_vector_type(16))) __bf16 v16bf;
typedef __attribute__((ext_vector_type(8)))  float  v8f;

union FragAB { v16bf v; unsigned u[8]; };
union FragC  { v8f   v; float    f[8]; };

// float -> bf16 (RNE) helpers (bit-level)
__device__ __forceinline__ unsigned short f2bf(float x) {
  unsigned a = __float_as_uint(x);
  return (unsigned short)((a + 0x7FFFu + ((a >> 16) & 1u)) >> 16);
}
__device__ __forceinline__ unsigned f2bf_pack(float lo, float hi) {
  return (unsigned)f2bf(lo) | ((unsigned)f2bf(hi) << 16);
}
__device__ __forceinline__ float bf2f(unsigned short u) {
  return __uint_as_float(((unsigned)u) << 16);
}

__device__ __forceinline__ unsigned hashu(unsigned a) {
  a ^= a >> 16; a *= 0x7feb352du; a ^= a >> 15; a *= 0x846ca68bu; a ^= a >> 16;
  return a;
}
__device__ __forceinline__ float u01(unsigned h) {
  return ((h >> 8) + 0.5f) * (1.0f / 16777216.0f);
}
__device__ __forceinline__ float sigf(float x) { return 1.0f / (1.0f + __expf(-x)); }

__device__ float digamma_dev(float x) {
  float r = 0.0f;
  #pragma unroll 1
  while (x < 6.0f) { r -= 1.0f / x; x += 1.0f; }
  float inv = 1.0f / x, inv2 = inv * inv;
  return r + __logf(x) - 0.5f * inv -
         inv2 * (1.0f / 12.0f - inv2 * (1.0f / 120.0f - inv2 * (1.0f / 252.0f)));
}

// gfx1250 async load: 16 bytes global -> LDS, tracked by ASYNCcnt
__device__ __forceinline__ void async_load_b128(unsigned lds_addr, const void* gptr) {
  asm volatile("global_load_async_to_lds_b128 %0, %1, off"
               :: "v"(lds_addr), "v"(gptr) : "memory");
}
__device__ __forceinline__ void wait_asynccnt0() {
  asm volatile("s_wait_asynccnt 0x0" ::: "memory");
}

// ---------------------------------------------------------------------------
// WMMA bf16 GEMM:  Cout[m, coff+n] = act( A[m, aoff+k] * W2[n, k] + bias[n] (+Cin) )
//   A : f32, row stride lda (converted to bf16 while staging to LDS)
//   W2: bf16 (u16), (N, Kd) row-major  -- pre-transposed weights so each
//       128x32 B tile is 128 rows of 64 contiguous bytes (async-copy friendly)
//   block tile 128x128, 8 waves (4 along M x 2 along N), each wave 2x4 WMMA tiles
// act: 0 none, 1 relu, 2 tanh
// ---------------------------------------------------------------------------
__global__ __launch_bounds__(256)
void gemm_bias_act(const float* __restrict__ A, int lda, long aoff,
                   const unsigned short* __restrict__ W2,
                   const float* __restrict__ bias,
                   const float* __restrict__ Cin,
                   float* __restrict__ Cout, int ldc, int coff,
                   int M, int N, int Kd, int act) {
  __shared__ unsigned As[128][17];   // [m][k_pair]  (2 bf16 per dword along K)
  __shared__ unsigned Bs[128 * 20];  // row n at 80B pitch (16B-aligned, bank-spread)

  const int tid  = threadIdx.x;
  const int lane = tid & 31;
  const int wave = tid >> 5;
  const int wm   = wave & 3;   // wave row   (32 rows each)
  const int wn   = wave >> 2;  // wave col   (64 cols each)
  const int m0   = blockIdx.y * 128;
  const int n0   = blockIdx.x * 128;
  const int half = lane >> 4;  // lane group 0/1
  const int lr   = lane & 15;

  const unsigned bs_base = (unsigned)(unsigned long long)(const void*)&Bs[0];

  FragC acc[2][4];
  #pragma unroll
  for (int i = 0; i < 2; i++)
    #pragma unroll
    for (int j = 0; j < 4; j++)
      #pragma unroll
      for (int r = 0; r < 8; r++) acc[i][j].f[r] = 0.0f;

  for (int k0 = 0; k0 < Kd; k0 += 32) {
    __syncthreads();
    // ---- async-stage B tile: 128 rows x 64B, one b128 chunk per lane ----
    #pragma unroll
    for (int i = 0; i < 2; i++) {
      int c  = tid + i * 256;          // 0..511
      int n  = c >> 2;
      int ch = c & 3;                  // 16B chunk within the 64B row
      int gn = n0 + n;
      if (gn < N) {
        const void* g = (const void*)(W2 + (long)gn * Kd + k0 + ch * 8);
        async_load_b128(bs_base + (unsigned)(n * 80 + ch * 16), g);
      } else {
        uint4 z = {0u, 0u, 0u, 0u};
        *(uint4*)&Bs[n * 20 + ch * 4] = z;   // zero-fill guard columns
      }
    }
    // ---- stage A tile (f32 -> packed bf16): 128 rows x 16 dwords ----
    #pragma unroll
    for (int i = 0; i < 8; i++) {
      int d  = tid + i * 256;          // 0..2047
      int m  = d >> 4;
      int kp = d & 15;
      const float* ap = A + (long)(m0 + m) * lda + aoff + k0 + kp * 2;
      As[m][kp] = f2bf_pack(ap[0], ap[1]);
      if (i == 0 && k0 + 32 < Kd) __builtin_prefetch(ap + 32, 0, 3);
    }
    wait_asynccnt0();
    __syncthreads();

    // ---- 2x4 WMMA tiles per wave ----
    #pragma unroll
    for (int ms = 0; ms < 2; ms++) {
      FragAB afrag;
      int m = wm * 32 + ms * 16 + lr;
      #pragma unroll
      for (int j = 0; j < 8; j++) {
        // ISA 7.12.2 16-bit A 16x32: j<4 -> K = half*8 + 2j ; j>=4 -> 16 + half*8 + 2(j-4)
        int k = ((j < 4) ? 0 : 16) + half * 8 + (j & 3) * 2;
        afrag.u[j] = As[m][k >> 1];
      }
      #pragma unroll
      for (int ns = 0; ns < 4; ns++) {
        FragAB bfrag;
        int n = wn * 64 + ns * 16 + lr;
        #pragma unroll
        for (int j = 0; j < 8; j++) {
          // B 32x16: j<4 -> K = half*8 + 2j ; j>=4 -> 16 + half*8 + 2(j-4)
          int k = ((j < 4) ? 0 : 16) + half * 8 + (j & 3) * 2;
          bfrag.u[j] = Bs[n * 20 + (k >> 1)];
        }
        acc[ms][ns].v = __builtin_amdgcn_wmma_f32_16x16x32_bf16(
            false, afrag.v, false, bfrag.v, (short)0, acc[ms][ns].v, false, false);
      }
    }
  }
  __syncthreads();

  // ---- epilogue: bias + optional addend + activation ----
  #pragma unroll
  for (int ms = 0; ms < 2; ms++) {
    #pragma unroll
    for (int ns = 0; ns < 4; ns++) {
      int gcol = n0 + wn * 64 + ns * 16 + lr;
      if (gcol >= N) continue;
      float bv = bias ? bias[gcol] : 0.0f;
      #pragma unroll
      for (int r = 0; r < 8; r++) {
        int grow = m0 + wm * 32 + ms * 16 + half * 8 + r;
        long idx = (long)grow * ldc + coff + gcol;
        float v = acc[ms][ns].f[r] + bv;
        if (Cin) v += Cin[idx];
        if (act == 1) v = v > 0.0f ? v : 0.0f;
        else if (act == 2) v = tanhf(v);
        Cout[idx] = v;
      }
    }
  }
}

// ---------------------------------------------------------------------------
// Weight conversion f32 -> bf16
//   cvt:  plain copy              (src already (N, Kd): dec LSTM weights)
//   cvtT: src (R, C) -> dst (C, R) (MLP weights stored (din, dout))
// ---------------------------------------------------------------------------
__global__ void cvt_bf16_kernel(unsigned short* dst, const float* src, int n) {
  for (int i = blockIdx.x * blockDim.x + threadIdx.x; i < n; i += gridDim.x * blockDim.x)
    dst[i] = f2bf(src[i]);
}
__global__ void cvt_bf16_t_kernel(unsigned short* dst, const float* src, int R, int C) {
  for (int i = blockIdx.x * blockDim.x + threadIdx.x; i < R * C; i += gridDim.x * blockDim.x) {
    int r = i / C, c = i % C;
    dst[(long)c * R + r] = f2bf(src[i]);
  }
}

// ---------------------------------------------------------------------------
// Row-wise softmax (+eps), cols <= 32, one wave per row
// ---------------------------------------------------------------------------
__global__ void softmax_eps_kernel(float* d, int rows, int cols, int ld, float eps) {
  int row  = blockIdx.x * (blockDim.x >> 5) + (threadIdx.x >> 5);
  int lane = threadIdx.x & 31;
  if (row >= rows) return;
  float x = (lane < cols) ? d[(long)row * ld + lane] : -3.0e38f;
  float m = x;
  #pragma unroll
  for (int o = 16; o > 0; o >>= 1) m = fmaxf(m, __shfl_xor(m, o, 32));
  float e = (lane < cols) ? __expf(x - m) : 0.0f;
  float s = e;
  #pragma unroll
  for (int o = 16; o > 0; o >>= 1) s += __shfl_xor(s, o, 32);
  if (lane < cols) d[(long)row * ld + lane] = e / s + eps;
}

// Dirichlet reparameterized sample: pi = g / (sum g + eps), g=(a*u*e^lgamma(a))^(1/a)
__global__ void dir_sample_kernel(const float* __restrict__ alpha, float* __restrict__ pi,
                                  int rows, unsigned seed) {
  int row  = blockIdx.x * (blockDim.x >> 5) + (threadIdx.x >> 5);
  int lane = threadIdx.x & 31;
  if (row >= rows) return;
  long idx = (long)row * 32 + lane;
  float a = alpha[idx];
  float u = u01(hashu(seed ^ (unsigned)(idx * 0x9E3779B9u + 0x3C6EF372u)));
  float g = powf(fmaxf(a * u * __expf(lgammaf(a)), 1e-30f), 1.0f / a);
  g = fminf(g, 1e30f);
  float s = g;
  #pragma unroll
  for (int o = 16; o > 0; o >>= 1) s += __shfl_xor(s, o, 32);
  pi[idx] = g / (s + 1e-6f);
}

// mvn2[l, k] = c_means[k, l] + exp(0.5*log_c_vars[k]) * N(0,1)
// stored transposed (L=256 rows, K=32 cols) so it feeds the async GEMM B path.
__global__ void mvn_gen_kernel(const float* __restrict__ cmu, const float* __restrict__ lcv,
                               unsigned short* __restrict__ mvn, unsigned seed) {
  int i = blockIdx.x * blockDim.x + threadIdx.x;
  if (i >= 32 * 256) return;
  int l = i >> 5, k = i & 31;
  unsigned h1 = hashu(seed ^ (unsigned)((k * 256 + l) * 0x85EBCA6Bu + 0x165667B1u));
  unsigned h2 = hashu(h1 ^ 0x27D4EB2Fu);
  float n = sqrtf(-2.0f * __logf(u01(h1))) * __cosf(6.28318530718f * u01(h2));
  mvn[i] = f2bf(cmu[k * 256 + l] + __expf(0.5f * lcv[k]) * n);
}

// est_z0 = (1/K) * sum_k mvn2[:,k]  broadcast over batch
__global__ void z0_init_kernel(const unsigned short* __restrict__ mvn, float* __restrict__ z) {
  int i = blockIdx.x * blockDim.x + threadIdx.x;
  if (i >= 2048 * 256) return;
  int l = i & 255;
  float s = 0.0f;
  #pragma unroll
  for (int k = 0; k < 32; k++) s += bf2f(mvn[l * 32 + k]);
  z[i] = s * (1.0f / 32.0f);
}

// LSTM gate update: z (B x 2048) -> h,c (B x 512); gate order i,f,g,o
__global__ void lstm_gates_kernel(const float* __restrict__ z, float* __restrict__ h,
                                  float* __restrict__ c) {
  int i = blockIdx.x * blockDim.x + threadIdx.x;
  if (i >= 2048 * 512) return;
  int b = i >> 9, j = i & 511;
  const float* zr = z + (long)b * 2048;
  float gi = zr[j], gf = zr[512 + j], gg = zr[1024 + j], go = zr[1536 + j];
  float cv = sigf(gf) * c[i] + sigf(gi) * tanhf(gg);
  h[i] = sigf(go) * tanhf(cv);
  c[i] = cv;
}

// loss += mean_b sum_i [ -0.5*(log 2pi + logvar + (x-mu)^2/var) ]
__global__ void log_gauss_kernel(const float* __restrict__ X, long xoff, int ldx,
                                 const float* __restrict__ o, float* __restrict__ loss) {
  __shared__ float red[256];
  float acc = 0.0f;
  for (int i = blockIdx.x * blockDim.x + threadIdx.x; i < 2048 * 256;
       i += gridDim.x * blockDim.x) {
    int b = i >> 8, ccol = i & 255;
    float xt  = X[(long)b * ldx + xoff + ccol];
    float mu  = o[(long)b * 512 + ccol];
    float lv  = o[(long)b * 512 + 256 + ccol];
    float var = __expf(lv) + 1e-6f;
    float d   = xt - mu;
    acc += -0.5f * (1.83787706641f + __logf(var) + d * d / var);
  }
  red[threadIdx.x] = acc;
  __syncthreads();
  #pragma unroll
  for (int s = 128; s > 0; s >>= 1) {
    if (threadIdx.x < s) red[threadIdx.x] += red[threadIdx.x + s];
    __syncthreads();
  }
  if (threadIdx.x == 0) atomicAdd(loss, red[0] * (1.0f / 2048.0f));
}

// loss -= mean_b KL(Dir(a1) || Dir(a2)), K=32, one wave per row
__global__ void dir_kl_kernel(const float* __restrict__ a1, const float* __restrict__ a2,
                              float* __restrict__ loss) {
  int row  = blockIdx.x * (blockDim.x >> 5) + (threadIdx.x >> 5);
  int lane = threadIdx.x & 31;
  if (row >= 2048) return;
  long idx = (long)row * 32 + lane;
  float x1 = a1[idx], x2 = a2[idx];
  float s1 = x1, s2 = x2;
  #pragma unroll
  for (int o = 16; o > 0; o >>= 1) { s1 += __shfl_xor(s1, o, 32); s2 += __shfl_xor(s2, o, 32); }
  float t = -(lgammaf(x1) - lgammaf(x2)) + (x1 - x2) * (digamma_dev(x1) - digamma_dev(s1));
  float ts = t;
  #pragma unroll
  for (int o = 16; o > 0; o >>= 1) ts += __shfl_xor(ts, o, 32);
  if (lane == 0)
    atomicAdd(loss, -(lgammaf(s1) - lgammaf(s2) + ts) * (1.0f / 2048.0f));
}

// loss += -(y * log(ypred + eps)).sum(-1).mean()
__global__ void pred_loss_kernel(const float* __restrict__ yp, const float* __restrict__ y,
                                 float* __restrict__ loss) {
  int row  = blockIdx.x * (blockDim.x >> 5) + (threadIdx.x >> 5);
  int lane = threadIdx.x & 31;
  if (row >= 2048) return;
  float t = (lane < 8) ? y[(long)row * 8 + lane] * __logf(yp[(long)row * 8 + lane] + 1e-6f)
                       : 0.0f;
  #pragma unroll
  for (int o = 16; o > 0; o >>= 1) t += __shfl_xor(t, o, 32);
  if (lane == 0) atomicAdd(loss, -t * (1.0f / 2048.0f));
}

// strided column-block copy (for concats / state init)
__global__ void copy_cols_kernel(float* __restrict__ dst, int dld, int doff,
                                 const float* __restrict__ src, int sld, int soff,
                                 int rows, int cols) {
  for (int i = blockIdx.x * blockDim.x + threadIdx.x; i < rows * cols;
       i += gridDim.x * blockDim.x) {
    int r = i / cols, c = i % cols;
    dst[(long)r * dld + doff + c] = src[(long)r * sld + soff + c];
  }
}

__global__ void finalize_kernel(const float* loss, float* out) { out[0] = -loss[0]; }

// ---------------------------------------------------------------------------
// Host orchestration
// ---------------------------------------------------------------------------
static inline void GEMM(hipStream_t s, const float* A, int lda, long aoff,
                        const unsigned short* W2, const float* bias, const float* Cin,
                        float* Cout, int ldc, int coff, int M, int N, int Kd, int act) {
  dim3 g((N + 127) / 128, M / 128);
  gemm_bias_act<<<g, dim3(256), 0, s>>>(A, lda, aoff, W2, bias, Cin, Cout, ldc, coff,
                                        M, N, Kd, act);
}

extern "C" void kernel_launch(void* const* d_in, const int* in_sizes, int n_in,
                              void* d_out, int out_size, void* d_ws, size_t ws_size,
                              hipStream_t stream) {
  // input order: x, y, then params flattened with sorted dict keys (jax pytree order)
  const float* X     = (const float*)d_in[0];   // (B,T,I)
  const float* Yt    = (const float*)d_in[1];   // (B,O)
  const float* cMean = (const float*)d_in[2];   // c_means (K,L)
  const float* dWhh  = (const float*)d_in[3];   // (4H,H)
  const float* dWih  = (const float*)d_in[4];   // (4H,DI)
  const float* dWo   = (const float*)d_in[5];   // (DI,H)
  const float* dbhh  = (const float*)d_in[6];
  const float* dbih  = (const float*)d_in[7];
  const float* dbo   = (const float*)d_in[8];
  const float* encOW = (const float*)d_in[9];   const float* encOb = (const float*)d_in[10];
  const float* encW0 = (const float*)d_in[11];  const float* encW1 = (const float*)d_in[12];
  const float* encb0 = (const float*)d_in[13];  const float* encb1 = (const float*)d_in[14];
  const float* logCV = (const float*)d_in[15];  // log_c_vars (K)
  const float* phxW0 = (const float*)d_in[16];  const float* phxW1 = (const float*)d_in[17];
  const float* phxb0 = (const float*)d_in[18];  const float* phxb1 = (const float*)d_in[19];
  const float* phxOW = (const float*)d_in[20];  const float* phxOb = (const float*)d_in[21];
  const float* phzW0 = (const float*)d_in[22];  const float* phzW1 = (const float*)d_in[23];
  const float* phzb0 = (const float*)d_in[24];  const float* phzb1 = (const float*)d_in[25];
  const float* phzOW = (const float*)d_in[26];  const float* phzOb = (const float*)d_in[27];
  const float* preOW = (const float*)d_in[28];  const float* preOb = (const float*)d_in[29];
  const float* prdW0 = (const float*)d_in[30];  const float* prdW1 = (const float*)d_in[31];
  const float* prdb0 = (const float*)d_in[32];  const float* prdb1 = (const float*)d_in[33];
  const float* priW0 = (const float*)d_in[34];  const float* priW1 = (const float*)d_in[35];
  const float* prib0 = (const float*)d_in[36];  const float* prib1 = (const float*)d_in[37];
  const float* priOW = (const float*)d_in[38];  const float* priOb = (const float*)d_in[39];
  const float* stW0  = (const float*)d_in[40];  const float* stW1  = (const float*)d_in[41];
  const float* stb0  = (const float*)d_in[42];  const float* stb1  = (const float*)d_in[43];
  const float* stOW  = (const float*)d_in[44];  const float* stOb  = (const float*)d_in[45];

  const int B = 2048;

  // ---- workspace carve-up ----
  char* base = (char*)d_ws;
  size_t cur = 0;
  auto allocf = [&](size_t n) -> float* {
    cur = (cur + 255) & ~(size_t)255; float* p = (float*)(base + cur); cur += n * 4; return p;
  };
  auto allocu = [&](size_t n) -> unsigned short* {
    cur = (cur + 255) & ~(size_t)255; auto* p = (unsigned short*)(base + cur); cur += n * 2; return p;
  };

  float* loss  = allocf(64);
  float* hbuf  = allocf((size_t)B * 256);
  float* est_z = allocf((size_t)B * 256);
  float* px    = allocf((size_t)B * 256);
  float* hid1  = allocf((size_t)B * 512);
  float* hid2  = allocf((size_t)B * 512);
  float* encin = allocf((size_t)B * 512);
  float* stin  = allocf((size_t)B * 768);
  float* ctx   = allocf((size_t)B * 512);
  float* h_l   = allocf((size_t)B * 512);
  float* c_l   = allocf((size_t)B * 512);
  float* o_t   = allocf((size_t)B * 512);
  float* zbuf  = allocf((size_t)B * 2048);
  float* aenc  = allocf((size_t)B * 32);
  float* aprio = allocf((size_t)B * 32);
  float* pibuf = allocf((size_t)B * 32);
  float* pred  = allocf((size_t)B * 8);

  // all weights stored (N, Kd) row-major bf16
  unsigned short* wEnc0 = allocu(512 * 512);  unsigned short* wEnc1 = allocu(512 * 512);
  unsigned short* wEncO = allocu(32 * 512);
  unsigned short* wPri0 = allocu(512 * 256);  unsigned short* wPri1 = allocu(512 * 512);
  unsigned short* wPriO = allocu(32 * 512);
  unsigned short* wPrd0 = allocu(512 * 256);  unsigned short* wPrd1 = allocu(512 * 512);
  unsigned short* wPreO = allocu(8 * 512);
  unsigned short* wPhx0 = allocu(512 * 256);  unsigned short* wPhx1 = allocu(512 * 512);
  unsigned short* wPhxO = allocu(256 * 512);
  unsigned short* wPhz0 = allocu(512 * 256);  unsigned short* wPhz1 = allocu(512 * 512);
  unsigned short* wPhzO = allocu(256 * 512);
  unsigned short* wSt0  = allocu(512 * 768);  unsigned short* wSt1  = allocu(512 * 512);
  unsigned short* wStO  = allocu(256 * 512);
  unsigned short* wWih  = allocu(2048 * 512); // stored (4H, DI) == (N, Kd) already
  unsigned short* wWhh  = allocu(2048 * 512);
  unsigned short* wWo   = allocu(512 * 512);
  unsigned short* mvnb  = allocu(256 * 32);   // (L, K)

  // ---- weight conversion (deterministic; redone every call) ----
  auto cvt  = [&](unsigned short* d, const float* s, int n) {
    cvt_bf16_kernel<<<dim3((n + 255) / 256), dim3(256), 0, stream>>>(d, s, n);
  };
  auto cvtT = [&](unsigned short* d, const float* s, int R, int C) {  // (R,C)->(C,R)
    cvt_bf16_t_kernel<<<dim3((R * C + 255) / 256), dim3(256), 0, stream>>>(d, s, R, C);
  };
  cvtT(wEnc0, encW0, 512, 512); cvtT(wEnc1, encW1, 512, 512); cvtT(wEncO, encOW, 512, 32);
  cvtT(wPri0, priW0, 256, 512); cvtT(wPri1, priW1, 512, 512); cvtT(wPriO, priOW, 512, 32);
  cvtT(wPrd0, prdW0, 256, 512); cvtT(wPrd1, prdW1, 512, 512); cvtT(wPreO, preOW, 512, 8);
  cvtT(wPhx0, phxW0, 256, 512); cvtT(wPhx1, phxW1, 512, 512); cvtT(wPhxO, phxOW, 512, 256);
  cvtT(wPhz0, phzW0, 256, 512); cvtT(wPhz1, phzW1, 512, 512); cvtT(wPhzO, phzOW, 512, 256);
  cvtT(wSt0,  stW0,  768, 512); cvtT(wSt1,  stW1,  512, 512); cvtT(wStO,  stOW,  512, 256);
  cvt(wWih, dWih, 2048 * 512);
  cvt(wWhh, dWhh, 2048 * 512);
  cvt(wWo,  dWo,  512 * 512);

  auto MLP3 = [&](const float* A, int lda, long aoff, int din,
                  const unsigned short* w0, const float* b0,
                  const unsigned short* w1, const float* b1,
                  const unsigned short* wO, const float* bO, int dout,
                  float* dst, int ldc, int coff, int outact) {
    GEMM(stream, A, lda, aoff, w0, b0, nullptr, hid1, 512, 0, B, 512, din, 1);
    GEMM(stream, hid1, 512, 0, w1, b1, nullptr, hid2, 512, 0, B, 512, 512, 0);
    GEMM(stream, hid2, 512, 0, wO, bO, nullptr, dst, ldc, coff, B, dout, 512, outact);
  };
  auto COPY = [&](float* d, int dld, int doff, const float* s, int sld, int soff, int cols) {
    copy_cols_kernel<<<dim3(1024), dim3(256), 0, stream>>>(d, dld, doff, s, sld, soff, B, cols);
  };

  // ---- init ----
  hipMemsetAsync(loss, 0, sizeof(float), stream);
  hipMemsetAsync(hbuf, 0, (size_t)B * 256 * 4, stream);
  mvn_gen_kernel<<<dim3(32), dim3(256), 0, stream>>>(cMean, logCV, mvnb, 0xC0FFEE01u);
  z0_init_kernel<<<dim3(2048), dim3(256), 0, stream>>>(mvnb, est_z);

  // ---- main sequential loop: 8 windows x 16 steps ----
  for (int w = 0; w < 8; w++) {
    // ctx = [phi_z(est_z), h]; LSTM h=c=ctx; i_0 = 0
    MLP3(est_z, 256, 0, 256, wPhz0, phzb0, wPhz1, phzb1, wPhzO, phzOb, 256, ctx, 512, 0, 2);
    COPY(ctx, 512, 256, hbuf, 256, 0, 256);
    COPY(h_l, 512, 0, ctx, 512, 0, 512);
    COPY(c_l, 512, 0, ctx, 512, 0, 512);
    hipMemsetAsync(o_t, 0, (size_t)B * 512 * 4, stream);

    for (int t = 0; t < 16; t++) {
      unsigned st = (unsigned)(w * 16 + t);
      long xoff = (long)(w * 16 + t) * 256;

      // decoder LSTM step (interleaved with t_step; mu/logvar = o_t halves)
      GEMM(stream, o_t, 512, 0, wWih, dbih, nullptr, zbuf, 2048, 0, B, 2048, 512, 0);
      GEMM(stream, h_l, 512, 0, wWhh, dbhh, zbuf, zbuf, 2048, 0, B, 2048, 512, 0);
      lstm_gates_kernel<<<dim3(4096), dim3(256), 0, stream>>>(zbuf, h_l, c_l);
      GEMM(stream, c_l, 512, 0, wWo, dbo, nullptr, o_t, 512, 0, B, 512, 512, 0);

      // a_prior = softmax(prior(h)) + eps
      MLP3(hbuf, 256, 0, 256, wPri0, prib0, wPri1, prib1, wPriO, priOb, 32, aprio, 32, 0, 0);
      softmax_eps_kernel<<<dim3(256), dim3(256), 0, stream>>>(aprio, B, 32, 32, 1e-6f);

      // px = phi_x(x_t)
      MLP3(X, 128 * 256, xoff, 256, wPhx0, phxb0, wPhx1, phxb1, wPhxO, phxOb, 256,
           px, 256, 0, 2);

      // a_enc = softmax(encoder([px, h])) + eps
      COPY(encin, 512, 0, px, 256, 0, 256);
      COPY(encin, 512, 256, hbuf, 256, 0, 256);
      MLP3(encin, 512, 0, 512, wEnc0, encb0, wEnc1, encb1, wEncO, encOb, 32, aenc, 32, 0, 0);
      softmax_eps_kernel<<<dim3(256), dim3(256), 0, stream>>>(aenc, B, 32, 32, 1e-6f);

      // pi = dir_sample(a_enc, u);  est_z = pi @ (c_means + sd*noise)
      dir_sample_kernel<<<dim3(256), dim3(256), 0, stream>>>(aenc, pibuf, B,
                                                             0xA5100000u + st);
      mvn_gen_kernel<<<dim3(32), dim3(256), 0, stream>>>(cMean, logCV, mvnb,
                                                         0x5EED0000u + st);
      GEMM(stream, pibuf, 32, 0, mvnb, nullptr, nullptr, est_z, 256, 0, B, 256, 32, 0);

      // loss += log_gauss(x_t; mu, var) - KL(Dir(a_enc)||Dir(a_prior))
      log_gauss_kernel<<<dim3(256), dim3(256), 0, stream>>>(X, xoff, 128 * 256, o_t, loss);
      dir_kl_kernel<<<dim3(256), dim3(256), 0, stream>>>(aenc, aprio, loss);

      // h = tanh(state_out(state([phi_z(est_z), px, h])))
      MLP3(est_z, 256, 0, 256, wPhz0, phzb0, wPhz1, phzb1, wPhzO, phzOb, 256,
           stin, 768, 0, 2);
      COPY(stin, 768, 256, px, 256, 0, 256);
      COPY(stin, 768, 512, hbuf, 256, 0, 256);
      MLP3(stin, 768, 0, 768, wSt0, stb0, wSt1, stb1, wStO, stOb, 256, hbuf, 256, 0, 2);
    }
  }

  // predictor head + cross-entropy
  MLP3(est_z, 256, 0, 256, wPrd0, prdb0, wPrd1, prdb1, wPreO, preOb, 8, pred, 8, 0, 0);
  softmax_eps_kernel<<<dim3(256), dim3(256), 0, stream>>>(pred, B, 8, 8, 0.0f);
  pred_loss_kernel<<<dim3(256), dim3(256), 0, stream>>>(pred, Yt, loss);

  finalize_kernel<<<dim3(1), dim3(1), 0, stream>>>(loss, (float*)d_out);
}
